// Attention_47588237640048
// MI455X (gfx1250) — compile-verified, matrix operand order checked by reading, not run on hardware
//
#include <hip/hip_runtime.h>

// ---------------------------------------------------------------------------
// Problem constants (from reference)
// ---------------------------------------------------------------------------
#define BATCH   2048
#define NTOK    64
#define CH      512
#define NH      16
#define HD      32
#define THREEC  1536
#define MROWS   (BATCH * NTOK)          // 131072
#define SCALE_F 0.17677669529663687f    // 32^-0.5
#define EPS_F   1e-5f

typedef _Float16 half_t;
typedef __attribute__((ext_vector_type(16))) _Float16 v16h;
typedef __attribute__((ext_vector_type(8)))  float    v8f;
typedef unsigned int v4u __attribute__((ext_vector_type(4)));
typedef int          v8i __attribute__((ext_vector_type(8)));
typedef int          v4i __attribute__((ext_vector_type(4)));

#if defined(__has_builtin)
#if __has_builtin(__builtin_amdgcn_tensor_load_to_lds) && \
    __has_builtin(__builtin_amdgcn_s_wait_tensorcnt)
#define USE_TDM 1
#endif
#endif
#ifndef USE_TDM
#define USE_TDM 0
#endif

// ---------------------------------------------------------------------------
// Kernel 1: LayerNorm over C=512, output f16. One wave32 per row.
// ---------------------------------------------------------------------------
__global__ __launch_bounds__(256) void ln_kernel(
    const float* __restrict__ x, const float* __restrict__ gamma,
    const float* __restrict__ beta, half_t* __restrict__ xn) {
  int row  = blockIdx.x * 8 + (threadIdx.x >> 5);
  int lane = threadIdx.x & 31;
  const float* xr = x + (size_t)row * CH + lane * 16;
  float v[16];
  float sum = 0.f, sq = 0.f;
#pragma unroll
  for (int j = 0; j < 16; j += 4) {
    float4 f = *(const float4*)&xr[j];
    v[j] = f.x; v[j + 1] = f.y; v[j + 2] = f.z; v[j + 3] = f.w;
    sum += f.x + f.y + f.z + f.w;
    sq  += f.x * f.x + f.y * f.y + f.z * f.z + f.w * f.w;
  }
#pragma unroll
  for (int m = 1; m < 32; m <<= 1) {
    sum += __shfl_xor(sum, m, 32);
    sq  += __shfl_xor(sq,  m, 32);
  }
  float mean = sum * (1.0f / CH);
  float var  = sq * (1.0f / CH) - mean * mean;
  float rstd = rsqrtf(var + EPS_F);
  v16h hv;
#pragma unroll
  for (int j = 0; j < 16; ++j) {
    int c = lane * 16 + j;
    hv[j] = (_Float16)((v[j] - mean) * rstd * gamma[c] + beta[c]);
  }
  *(v16h*)&xn[(size_t)row * CH + lane * 16] = hv;
}

// ---------------------------------------------------------------------------
// Kernel 2: transpose + convert weight f32[K][N] -> f16[N][K]
// ---------------------------------------------------------------------------
__global__ __launch_bounds__(256) void wt_kernel(
    const float* __restrict__ w, half_t* __restrict__ wT, int K, int N) {
  int t = blockIdx.x * 256 + threadIdx.x;     // over K*N
  int k = t / N, n = t - k * N;
  wT[(size_t)n * K + k] = (half_t)w[t];
}

// ---------------------------------------------------------------------------
// Kernel 3: gather relative-position bias -> biasHIJ[h][i][j] (f32)
// ---------------------------------------------------------------------------
__global__ __launch_bounds__(256) void bias_kernel(
    const float* __restrict__ table, const int* __restrict__ relIdx,
    float* __restrict__ biasHIJ) {
  int t  = blockIdx.x * 256 + threadIdx.x;    // over NH*64*64
  int h  = t >> 12;
  int ij = t & 4095;
  biasHIJ[t] = table[(size_t)relIdx[ij] * NH + h];
}

// ---------------------------------------------------------------------------
// TDM helper: 2D f16 tile load, tile_d0 x tile_d1 elements, row stride
// stride_d0 elements, into LDS at byte offset lds_off.
// D# packing per CDNA5 ISA 8.3/8.4 (Group0: count/lds/global/type;
// Group1: data_size, tensor dims, tile dims, dim0 stride).
// ---------------------------------------------------------------------------
#if USE_TDM
__device__ __forceinline__ void tdm_load_tile_f16(
    unsigned int lds_off, const void* gaddr, unsigned int tensor_d0,
    unsigned int tensor_d1, unsigned int stride_d0, unsigned int tile_d0,
    unsigned int tile_d1) {
  unsigned long long ga = (unsigned long long)(uintptr_t)gaddr;
  v4u g0;
  g0[0] = 1u;                                        // count=1, user mode
  g0[1] = lds_off;                                   // lds_addr [63:32]
  g0[2] = (unsigned int)(ga & 0xffffffffull);        // global_addr [95:64]
  g0[3] = (unsigned int)((ga >> 32) & 0x1ffffffull)  // global_addr [120:96]
          | (2u << 30);                              // type=2 ("image")
  v8i g1;
  g1[0] = (int)(1u << 16);                           // data_size=1 -> 2 bytes
  g1[1] = (int)((tensor_d0 & 0xffffu) << 16);        // tensor_dim0[15:0]
  g1[2] = (int)((tensor_d0 >> 16) | ((tensor_d1 & 0xffffu) << 16));
  g1[3] = (int)((tensor_d1 >> 16) | (tile_d0 << 16));// tile_dim0
  g1[4] = (int)tile_d1;                              // tile_dim1 (tile_dim2=0)
  g1[5] = (int)stride_d0;                            // tensor_dim0_stride[31:0]
  g1[6] = 0;                                         // stride msbs / dim1_stride
  g1[7] = 0;
  v4i z4 = (v4i){0, 0, 0, 0};
#if __clang_major__ >= 23
  v8i z8 = (v8i){0, 0, 0, 0, 0, 0, 0, 0};
  __builtin_amdgcn_tensor_load_to_lds(g0, g1, z4, z4, z8, 0);
#else
  __builtin_amdgcn_tensor_load_to_lds(g0, g1, z4, z4, 0);
#endif
}
#endif

// ---------------------------------------------------------------------------
// Tiled WMMA GEMM: C[M][N] = A[M][K](f16) * Bt[N][K](f16)^T + bias[N]
// Block tile 128x128, K-step 32, 8 waves, wave tile 32(M)x64(N).
// TDM path: double-buffered LDS tiles DMA'd by the Tensor Data Mover
// (issued by wave 0, retired with s_wait_tensorcnt), overlapping the copy
// of tile i+1 with the WMMAs on tile i.
// ---------------------------------------------------------------------------
template <bool OUT_F16>
__global__ __launch_bounds__(256) void gemm_kernel(
    const half_t* __restrict__ A, const half_t* __restrict__ Bt,
    const float* __restrict__ bias, void* __restrict__ Cout,
    int M, int N, int K) {
  __shared__ _Float16 As[2][128 * 32];
  __shared__ _Float16 Bs[2][128 * 32];

  const int tid   = threadIdx.x;
  const int lane  = tid & 31;
  const int wid   = tid >> 5;
  const int waveM = wid & 3;        // 4 strips of 32 rows
  const int waveN = wid >> 2;       // 2 strips of 64 cols
  const int l15   = lane & 15;
  const int hi    = lane >> 4;
  const int mBase = blockIdx.y * 128;
  const int nBase = blockIdx.x * 128;

  v8f acc[2][4];
#pragma unroll
  for (int mi = 0; mi < 2; ++mi)
#pragma unroll
    for (int ni = 0; ni < 4; ++ni) acc[mi][ni] = (v8f){};

#if USE_TDM
  const unsigned int ldsA0 = (unsigned int)(uintptr_t)(&As[0][0]);
  const unsigned int ldsA1 = (unsigned int)(uintptr_t)(&As[1][0]);
  const unsigned int ldsB0 = (unsigned int)(uintptr_t)(&Bs[0][0]);
  const unsigned int ldsB1 = (unsigned int)(uintptr_t)(&Bs[1][0]);
  if (wid == 0) {  // prologue: DMA tile 0 into buffer 0
    tdm_load_tile_f16(ldsA0, A + (size_t)mBase * K, (unsigned)K, (unsigned)M,
                      (unsigned)K, 32u, 128u);
    tdm_load_tile_f16(ldsB0, Bt + (size_t)nBase * K, (unsigned)K, (unsigned)N,
                      (unsigned)K, 32u, 128u);
  }
  int cur = 0;
  for (int kk = 0; kk < K; kk += 32) {
    const bool hasNext = (kk + 32) < K;
    if (wid == 0) {
      if (hasNext) {  // DMA next tile into the other buffer, then retire oldest
        unsigned int la = cur ? ldsA0 : ldsA1;
        unsigned int lb = cur ? ldsB0 : ldsB1;
        tdm_load_tile_f16(la, A + (size_t)mBase * K + kk + 32, (unsigned)K,
                          (unsigned)M, (unsigned)K, 32u, 128u);
        tdm_load_tile_f16(lb, Bt + (size_t)nBase * K + kk + 32, (unsigned)K,
                          (unsigned)N, (unsigned)K, 32u, 128u);
        __builtin_amdgcn_s_wait_tensorcnt(2);  // current A+B tile landed
      } else {
        __builtin_amdgcn_s_wait_tensorcnt(0);  // last tile landed
      }
    }
    __syncthreads();

    v16h af[2], bf[4];
#pragma unroll
    for (int mi = 0; mi < 2; ++mi)
      af[mi] =
          *(const v16h*)&As[cur][(waveM * 32 + mi * 16 + l15) * 32 + hi * 16];
#pragma unroll
    for (int ni = 0; ni < 4; ++ni)
      bf[ni] =
          *(const v16h*)&Bs[cur][(waveN * 64 + ni * 16 + l15) * 32 + hi * 16];

#pragma unroll
    for (int mi = 0; mi < 2; ++mi)
#pragma unroll
      for (int ni = 0; ni < 4; ++ni)
        acc[mi][ni] = __builtin_amdgcn_wmma_f32_16x16x32_f16(
            false, af[mi], false, bf[ni], (short)0, acc[mi][ni], false, false);
    __syncthreads();
    cur ^= 1;
  }
#else
  for (int kk = 0; kk < K; kk += 32) {
    // Fallback: stage A and B tiles into LDS (16B chunks, fully coalesced).
#pragma unroll
    for (int i = 0; i < 2; ++i) {
      int c = tid + i * 256;        // 512 chunks per tile
      int r = c >> 2, p = (c & 3) * 8;
      *(uint4*)&As[0][r * 32 + p] =
          *(const uint4*)&A[(size_t)(mBase + r) * K + kk + p];
      *(uint4*)&Bs[0][r * 32 + p] =
          *(const uint4*)&Bt[(size_t)(nBase + r) * K + kk + p];
    }
    if (kk + 32 < K) {
      __builtin_prefetch(&A[(size_t)(mBase + (tid >> 1)) * K + kk + 32], 0, 0);
      __builtin_prefetch(&Bt[(size_t)(nBase + (tid >> 1)) * K + kk + 32], 0, 0);
    }
    __syncthreads();

    v16h af[2], bf[4];
#pragma unroll
    for (int mi = 0; mi < 2; ++mi)
      af[mi] =
          *(const v16h*)&As[0][(waveM * 32 + mi * 16 + l15) * 32 + hi * 16];
#pragma unroll
    for (int ni = 0; ni < 4; ++ni)
      bf[ni] =
          *(const v16h*)&Bs[0][(waveN * 64 + ni * 16 + l15) * 32 + hi * 16];

#pragma unroll
    for (int mi = 0; mi < 2; ++mi)
#pragma unroll
      for (int ni = 0; ni < 4; ++ni)
        acc[mi][ni] = __builtin_amdgcn_wmma_f32_16x16x32_f16(
            false, af[mi], false, bf[ni], (short)0, acc[mi][ni], false, false);
    __syncthreads();
  }
#endif

  // Epilogue: add bias, store.
#pragma unroll
  for (int mi = 0; mi < 2; ++mi)
#pragma unroll
    for (int ni = 0; ni < 4; ++ni) {
      int col = nBase + waveN * 64 + ni * 16 + l15;
      float bv = bias[col];
#pragma unroll
      for (int r = 0; r < 8; ++r) {
        int row = mBase + waveM * 32 + mi * 16 + r + hi * 8;
        float val = acc[mi][ni][r] + bv;
        if (OUT_F16)
          ((half_t*)Cout)[(size_t)row * N + col] = (half_t)val;
        else
          ((float*)Cout)[(size_t)row * N + col] = val;
      }
    }
}

// ---------------------------------------------------------------------------
// Kernel 4: windowed attention. One workgroup per (b,h), 4 waves.
// qkv f16 [M][1536]: q at +0, k at +512, v at +1024, channel = h*32+d.
// out f16 [M][512], channel = h*32+d (ready for projection GEMM).
// ---------------------------------------------------------------------------
__global__ __launch_bounds__(128) void attn_kernel(
    const half_t* __restrict__ qkv, const float* __restrict__ biasHIJ,
    half_t* __restrict__ out) {
  const int bh = blockIdx.x;
  const int b  = bh >> 4;
  const int h  = bh & 15;
  const int tid  = threadIdx.x;
  const int lane = tid & 31;
  const int w    = tid >> 5;
  const int l15  = lane & 15;
  const int hi   = lane >> 4;
  const size_t baseTok = (size_t)b * NTOK;

  __shared__ _Float16 vT[HD * NTOK];       // [d][token]
  __shared__ _Float16 Pst[4][16 * NTOK];   // per-wave P strip [16][64]

  // Cooperatively transpose V into LDS.
  {
    int token = tid >> 1;
    int dh    = (tid & 1) * 16;
    const half_t* vp = qkv + (baseTok + token) * THREEC + 1024 + h * HD + dh;
#pragma unroll
    for (int d = 0; d < 16; ++d) vT[(dh + d) * NTOK + token] = vp[d];
  }
  __syncthreads();

  const int i0 = w * 16;  // 16-row strip per wave

  // S strip = q(16x32) x k^T(32x64): A and B frags load directly from global
  // (each lane reads 16 contiguous k, k = 16*hi + s, same order for A and B).
  v16h aq = *(const v16h*)(qkv + (baseTok + i0 + l15) * THREEC + h * HD + hi * 16);
  v8f s[4];
#pragma unroll
  for (int jt = 0; jt < 4; ++jt) {
    v16h bk = *(const v16h*)(qkv + (baseTok + jt * 16 + l15) * THREEC + 512 +
                             h * HD + hi * 16);
    v8f z = (v8f){};
    s[jt] = __builtin_amdgcn_wmma_f32_16x16x32_f16(false, aq, false, bk,
                                                   (short)0, z, false, false);
  }

  // Scale + bias + row softmax (rows live in 16-lane halves -> shfl_xor<16).
  const float* bp = biasHIJ + h * 4096;
  float p[4][8];
#pragma unroll
  for (int r = 0; r < 8; ++r) {
    int i = i0 + r + hi * 8;
    float mx = -1e30f;
#pragma unroll
    for (int jt = 0; jt < 4; ++jt) {
      int j = jt * 16 + l15;
      float t = s[jt][r] * SCALE_F + bp[i * 64 + j];
      p[jt][r] = t;
      mx = fmaxf(mx, t);
    }
#pragma unroll
    for (int m = 1; m < 16; m <<= 1) mx = fmaxf(mx, __shfl_xor(mx, m, 32));
    float sum = 0.f;
#pragma unroll
    for (int jt = 0; jt < 4; ++jt) {
      float e = __expf(p[jt][r] - mx);
      p[jt][r] = e;
      sum += e;
    }
#pragma unroll
    for (int m = 1; m < 16; m <<= 1) sum += __shfl_xor(sum, m, 32);
    float inv = 1.f / sum;
#pragma unroll
    for (int jt = 0; jt < 4; ++jt) p[jt][r] *= inv;
  }

  // Stage P (f16) through LDS to get the A-side fragment layout.
#pragma unroll
  for (int jt = 0; jt < 4; ++jt)
#pragma unroll
    for (int r = 0; r < 8; ++r)
      Pst[w][(r + hi * 8) * NTOK + jt * 16 + l15] = (_Float16)p[jt][r];
  __syncthreads();

  // O strip = P(16x64) x v(64x32): two chained K=32 WMMAs per 16-col tile.
  v8f o[2] = {(v8f){}, (v8f){}};
#pragma unroll
  for (int kc = 0; kc < 64; kc += 32) {
    v16h ap = *(const v16h*)&Pst[w][l15 * NTOK + kc + hi * 16];
#pragma unroll
    for (int nt = 0; nt < 2; ++nt) {
      v16h bv = *(const v16h*)&vT[(nt * 16 + l15) * NTOK + kc + hi * 16];
      o[nt] = __builtin_amdgcn_wmma_f32_16x16x32_f16(false, ap, false, bv,
                                                     (short)0, o[nt], false, false);
    }
  }

  // Store O strip, channel-major layout for the projection GEMM.
#pragma unroll
  for (int nt = 0; nt < 2; ++nt)
#pragma unroll
    for (int r = 0; r < 8; ++r) {
      int token = i0 + r + hi * 8;
      out[(baseTok + token) * CH + h * HD + nt * 16 + l15] = (half_t)o[nt][r];
    }
}

// ---------------------------------------------------------------------------
// Launcher
// ---------------------------------------------------------------------------
extern "C" void kernel_launch(void* const* d_in, const int* in_sizes, int n_in,
                              void* d_out, int out_size, void* d_ws, size_t ws_size,
                              hipStream_t stream) {
  const float* x        = (const float*)d_in[0];
  const float* ln_gamma = (const float*)d_in[1];
  const float* ln_beta  = (const float*)d_in[2];
  const float* qkv_w    = (const float*)d_in[3];
  const float* qkv_b    = (const float*)d_in[4];
  const float* proj_w   = (const float*)d_in[5];
  const float* proj_b   = (const float*)d_in[6];
  const float* btable   = (const float*)d_in[7];
  const int*   rel_idx  = (const int*)d_in[8];
  float* outp = (float*)d_out;

  // Workspace carve-up (256B aligned). attnOut reuses the xn region.
  char* w = (char*)d_ws;
  half_t* xn      = (half_t*)w;                              // 134,217,728 B
  half_t* attnOut = xn;                                      // reuse after qkv GEMM
  w += (size_t)MROWS * CH * sizeof(half_t);
  half_t* qkvBuf  = (half_t*)w;                              // 402,653,184 B
  w += (size_t)MROWS * THREEC * sizeof(half_t);
  half_t* qkvWT   = (half_t*)w;                              // 1,572,864 B
  w += (size_t)THREEC * CH * sizeof(half_t);
  half_t* projWT  = (half_t*)w;                              // 524,288 B
  w += (size_t)CH * CH * sizeof(half_t);
  float*  biasHIJ = (float*)w;                               // 262,144 B

  // 1) LayerNorm -> f16
  ln_kernel<<<MROWS / 8, 256, 0, stream>>>(x, ln_gamma, ln_beta, xn);
  // 2) Weight transpose/convert (one-shot)
  wt_kernel<<<(CH * THREEC) / 256, 256, 0, stream>>>(qkv_w, qkvWT, CH, THREEC);
  wt_kernel<<<(CH * CH) / 256, 256, 0, stream>>>(proj_w, projWT, CH, CH);
  // 3) Relative-position bias gather
  bias_kernel<<<(NH * NTOK * NTOK) / 256, 256, 0, stream>>>(btable, rel_idx, biasHIJ);
  // 4) QKV GEMM (f16 out)
  gemm_kernel<true><<<dim3(THREEC / 128, MROWS / 128), 256, 0, stream>>>(
      xn, qkvWT, qkv_b, (void*)qkvBuf, MROWS, THREEC, CH);
  // 5) Attention (one WG per (b,h))
  attn_kernel<<<BATCH * NH, 128, 0, stream>>>(qkvBuf, biasHIJ, attnOut);
  // 6) Projection GEMM (f32 out + bias)
  gemm_kernel<false><<<dim3(CH / 128, MROWS / 128), 256, 0, stream>>>(
      attnOut, projWT, proj_b, (void*)outp, MROWS, CH, CH);
}
